// RobustAngleSO3Distribution_83786222011231
// MI455X (gfx1250) — matrix operand order repaired and checked by reading, not compiled, not run on hardware
//
#include <hip/hip_runtime.h>
#include <hip/hip_bf16.h>
#include <stdint.h>

// ---------------------------------------------------------------------------
// RobustAngleSO3Distribution for MI455X (gfx1250, wave32, WMMA bf16)
//
//   probs[n,b] = c0[b] * sum_l exp(-l(l+1) sigma_n^2) * c2[b,l]
//   angle[n]   = Gumbel-max categorical over probs row + uniform jitter,
//                Gaussian fallback for sigma < 0.004
//
// Fused GEMM (65536 x 1024pad, K=1024pad) + sampling epilogue.
// A (c1) generated on the fly into LDS; B (c2^T) pre-swizzled once per launch
// into WMMA B-fragment layout (2 MB, L2-resident on 192 MB L2).
// Each wave runs TWO independent accumulator chains (adjacent bin tiles) for
// XDL back-to-back issue and 2x A-fragment reuse.
// ---------------------------------------------------------------------------

#define NROWS      65536
#define NBINS      1000
#define NL         1001
#define BT_TILES   64            // bins padded to 1024 (tiles 62.5 -> 64, pad = zero)
#define BT_PAIRS   32
#define KT_TILES   32            // K padded to 1024, 32 per WMMA
#define NBINS_PAD  1024
#define K_PAD      1024
#define A_STRIDE   1032          // 16B-aligned, breaks LDS bank conflicts
#define PI_D       3.14159265358979323846
#define BIN_W      (3.14159265358979323846 / 1000.0)
#define SIGMA_TH   0.004f
#define NEG_BIG    (-3.0e38f)

// B fragment pool: [bt][kt][lane][16 bf16] -> 64*32*32*16 elements (2 MB)
#define BFRAG_ELEMS ((size_t)BT_TILES * KT_TILES * 32 * 16)
#define BFRAG_BYTES (BFRAG_ELEMS * 2)

typedef __attribute__((ext_vector_type(16))) __bf16 v16bf;
typedef __attribute__((ext_vector_type(8)))  __bf16 v8bf;
typedef __attribute__((ext_vector_type(8)))  float  v8f;

union AFrag { v16bf v; v8bf h[2]; };

__device__ __forceinline__ __bf16 to_bf16(float f) {
    union { float f; uint32_t u; } v; v.f = f;
    uint32_t r = v.u + 0x7FFFu + ((v.u >> 16) & 1u);   // round-to-nearest-even
    union { uint16_t u; __bf16 b; } o; o.u = (uint16_t)(r >> 16);
    return o.b;
}

__device__ __forceinline__ uint32_t hashu(uint32_t x) {
    x ^= x >> 17; x *= 0xED5AD4BBu;
    x ^= x >> 11; x *= 0xAC4C1B51u;
    x ^= x >> 15; x *= 0x31848BABu;
    x ^= x >> 14;
    return x;
}
__device__ __forceinline__ float u01(uint32_t h) {
    return (float)(h >> 8) * (1.0f / 16777216.0f) + (0.5f / 16777216.0f);
}

// ---------------------------------------------------------------------------
// Setup kernel 1: c2 in bf16, pre-swizzled into the per-lane-contiguous WMMA
// B-fragment layout (lane = column (lane&15) and K-half (lane>>4); each
// lane's 16 consecutive-K bf16 values are contiguous -> single 32B load).
// Padded bins (b >= 1000) and padded K (l >= 1001) are zero.
// ---------------------------------------------------------------------------
__global__ __launch_bounds__(256) void build_c2_kernel(__bf16* __restrict__ bfrag) {
    uint32_t idx = blockIdx.x * 256u + threadIdx.x;
    if (idx >= (uint32_t)BFRAG_ELEMS) return;
    int e    = idx & 15;
    int lane = (idx >> 4) & 31;
    int kt   = (idx >> 9) & 31;
    int bt   = idx >> 14;
    int b = bt * 16 + (lane & 15);
    int l = kt * 32 + (lane >> 4) * 16 + e;
    float val = 0.f;
    if (b < NBINS && l < NL) {
        double bin = ((double)b + 0.5) * BIN_W;
        double s   = sin(bin * 0.5);
        val = (float)((2.0 * l + 1.0) * sin(((double)l + 0.5) * bin) / s);
    }
    bfrag[idx] = to_bf16(val);
}

// Setup kernel 2: c0[b] (zero in padded region -> never sampled) and centers.
__global__ __launch_bounds__(256) void build_tables_kernel(float* __restrict__ c0,
                                                           float* __restrict__ bins) {
    int b = blockIdx.x * 256 + threadIdx.x;
    if (b >= NBINS_PAD) return;
    double bin = ((double)b + 0.5) * BIN_W;
    bins[b] = (float)bin;
    c0[b]   = (b < NBINS) ? (float)((1.0 - cos(bin)) / PI_D) : 0.f;
}

// ---------------------------------------------------------------------------
// Main fused kernel: one workgroup = 16 sigma rows, 4 waves split 32 bin-tile
// PAIRS. K loop = 32 x 2 x v_wmma_f32_16x16x32_bf16 per pair (two independent
// accumulator chains). Epilogue is branchless Gumbel-max sampling.
// ---------------------------------------------------------------------------
__global__ __launch_bounds__(128) void so3_sample_kernel(
    const float* __restrict__ sigma,
    const __bf16* __restrict__ bfrag,
    const float* __restrict__ c0g,
    const float* __restrict__ binsg,
    float* __restrict__ out)
{
    __shared__ __bf16 Alds[16 * A_STRIDE];      // c1 strip, bf16
    __shared__ float  s2[16], sg[16];
    __shared__ float  c0s[NBINS_PAD];
    __shared__ float  binss[NBINS_PAD];
    __shared__ float  redS[4][16];
    __shared__ int    redB[4][16];

    const int tid     = threadIdx.x;
    const int rowbase = blockIdx.x * 16;

    if (tid < 16) {
        float s = sigma[rowbase + tid];
        sg[tid] = s;
        s2[tid] = s * s;
    }
    for (int i = tid; i < NBINS_PAD; i += 128) {
        c0s[i]   = c0g[i];
        binss[i] = binsg[i];
    }
    __syncthreads();

    // Generate A strip: c1[r,l] = exp(-l(l+1) sigma_r^2), zero-padded K.
    for (int i = tid; i < 16 * K_PAD; i += 128) {
        int r = i >> 10;
        int l = i & (K_PAD - 1);
        float v = 0.f;
        if (l < NL) v = __expf(-(float)l * (float)(l + 1) * s2[r]);
        Alds[r * A_STRIDE + l] = to_bf16(v);
    }
    __syncthreads();

    const int wave = tid >> 5;
    const int lane = tid & 31;
    const int col  = lane & 15;
    const int half = lane >> 4;

    float bscore[8];
    int   bbin[8];
#pragma unroll
    for (int j = 0; j < 8; ++j) { bscore[j] = NEG_BIG; bbin[j] = 0; }

    // A fragment base for this lane: row = lane&15, K-subchunk = (lane>>4)*16.
    const __bf16* arow = &Alds[col * A_STRIDE + half * 16];

    for (int pr = wave; pr < BT_PAIRS; pr += 4) {
        const int bt0 = pr * 2;          // adjacent tiles -> B data contiguous 64 KB
        v8f acc0 = {0.f, 0.f, 0.f, 0.f, 0.f, 0.f, 0.f, 0.f};
        v8f acc1 = {0.f, 0.f, 0.f, 0.f, 0.f, 0.f, 0.f, 0.f};

        // frag(bt,kt) at elements ((bt*32 + kt)*32 + lane) * 16
        const __bf16* bbase = bfrag + ((((size_t)bt0 * KT_TILES) * 32 + lane) << 4);
        const size_t  tstep = (size_t)KT_TILES * 32 * 16;   // bt -> bt+1

#pragma unroll 2
        for (int kt = 0; kt < KT_TILES; ++kt) {
            AFrag a;
            a.h[0] = *(const v8bf*)(arow + kt * 32);
            a.h[1] = *(const v8bf*)(arow + kt * 32 + 8);
            v16bf bv0 = *(const v16bf*)(bbase + ((size_t)kt << 9));
            v16bf bv1 = *(const v16bf*)(bbase + tstep + ((size_t)kt << 9));
            if (kt + 1 < KT_TILES) {
                __builtin_prefetch(bbase + ((size_t)(kt + 1) << 9), 0, 1);
                __builtin_prefetch(bbase + tstep + ((size_t)(kt + 1) << 9), 0, 1);
            }
            acc0 = __builtin_amdgcn_wmma_f32_16x16x32_bf16(
                false, a.v, false, bv0, (short)0, acc0, false, false);
            acc1 = __builtin_amdgcn_wmma_f32_16x16x32_bf16(
                false, a.v, false, bv1, (short)0, acc1, false, false);
        }

        // Branchless Gumbel-max epilogue over both 16x16 tiles.
        const int   b0  = bt0 * 16 + col;
        const int   b1  = b0 + 16;
        const float c00 = c0s[b0];
        const float c01 = c0s[b1];
#pragma unroll
        for (int j = 0; j < 8; ++j) {
            const int n = rowbase + j + half * 8;

            float p0 = c00 * acc0[j];
            float g0 = -__logf(-__logf(u01(hashu((uint32_t)n * 1009u + (uint32_t)b0 + 0x9E3779B9u))));
            float s0 = __logf(fmaxf(p0, 1e-37f)) + g0;
            s0 = (p0 > 0.f) ? s0 : NEG_BIG;
            bbin[j]   = (s0 > bscore[j]) ? b0 : bbin[j];
            bscore[j] = fmaxf(s0, bscore[j]);

            float p1 = c01 * acc1[j];
            float g1 = -__logf(-__logf(u01(hashu((uint32_t)n * 1009u + (uint32_t)b1 + 0x9E3779B9u))));
            float s1 = __logf(fmaxf(p1, 1e-37f)) + g1;
            s1 = (p1 > 0.f) ? s1 : NEG_BIG;
            bbin[j]   = (s1 > bscore[j]) ? b1 : bbin[j];
            bscore[j] = fmaxf(s1, bscore[j]);
        }
    }

    // Reduce across the 16 lanes of each C-matrix half (rows j / j+8).
#pragma unroll
    for (int j = 0; j < 8; ++j) {
        float s  = bscore[j];
        int   bb = bbin[j];
        for (int m = 1; m < 16; m <<= 1) {
            float os = __shfl_xor(s, m, 32);
            int   ob = __shfl_xor(bb, m, 32);
            bb = (os > s) ? ob : bb;
            s  = fmaxf(os, s);
        }
        bscore[j] = s; bbin[j] = bb;
    }
    if (lane == 0 || lane == 16) {
#pragma unroll
        for (int j = 0; j < 8; ++j) {
            redS[wave][half * 8 + j] = bscore[j];
            redB[wave][half * 8 + j] = bbin[j];
        }
    }
    __syncthreads();

    // Final cross-wave reduction + jitter + Gaussian fallback, 16 rows.
    if (tid < 16) {
        int r = tid;
        float s  = redS[0][r];
        int   bb = redB[0][r];
#pragma unroll
        for (int w = 1; w < 4; ++w) {
            bb = (redS[w][r] > s) ? redB[w][r] : bb;
            s  = fmaxf(redS[w][r], s);
        }

        int   n   = rowbase + r;
        float sig = sg[r];

        float u     = u01(hashu((uint32_t)n ^ 0xA511E9B3u));
        float angle = binss[bb] + (float)BIN_W * (u - 0.5f);

        if (sig < SIGMA_TH) {
            float u1 = u01(hashu((uint32_t)n * 2654435761u + 0x85EBCA6Bu));
            float u2 = u01(hashu((uint32_t)n * 40503u + 0xC2B2AE35u));
            float nrm = __fsqrt_rn(-2.f * __logf(u1)) * __cosf(6.28318530718f * u2);
            angle = sig * 2.f + sig * nrm;
        }
        out[n] = angle;
    }
}

// ---------------------------------------------------------------------------
extern "C" void kernel_launch(void* const* d_in, const int* in_sizes, int n_in,
                              void* d_out, int out_size, void* d_ws, size_t ws_size,
                              hipStream_t stream) {
    (void)in_sizes; (void)n_in; (void)out_size; (void)ws_size;

    const float* sigma = (const float*)d_in[0];
    float*       out   = (float*)d_out;

    char*   ws    = (char*)d_ws;
    __bf16* bfrag = (__bf16*)ws;
    float*  c0g   = (float*)(ws + BFRAG_BYTES);
    float*  binsg = (float*)(ws + BFRAG_BYTES + NBINS_PAD * sizeof(float));

    // Setup: constant tables (deterministic, regenerated every call).
    {
        uint32_t total = (uint32_t)BFRAG_ELEMS;
        build_c2_kernel<<<(total + 255) / 256, 256, 0, stream>>>(bfrag);
        build_tables_kernel<<<(NBINS_PAD + 255) / 256, 256, 0, stream>>>(c0g, binsg);
    }

    // Fused GEMM + categorical sampling: 4096 workgroups x 128 threads.
    so3_sample_kernel<<<NROWS / 16, 128, 0, stream>>>(sigma, bfrag, c0g, binsg, out);
}
